// VVITv3_4166118277491
// MI455X (gfx1250) — matrix operand order repaired and checked by reading, not compiled, not run on hardware
//
#include <hip/hip_runtime.h>

// ---------------------------------------------------------------------------
// Types / helpers
// ---------------------------------------------------------------------------
typedef __bf16 v16bf __attribute__((ext_vector_type(16)));
typedef __bf16 v8bf  __attribute__((ext_vector_type(8)));
typedef float  v8f   __attribute__((ext_vector_type(8)));
// Matches the async-copy builtin's expected pointer types (GCC vector style).
typedef int v4i_vs __attribute__((__vector_size__(4 * sizeof(int))));

#ifndef __has_builtin
#define __has_builtin(x) 0
#endif

// Async global->LDS copy path (gfx1250 GLOBAL_LOAD_ASYNC_TO_LDS_B128,
// tracked with ASYNCcnt). Guarded: falls back to register-relayed copy if the
// toolchain doesn't declare the builtins (also keeps the host pass clean).
#if defined(__HIP_DEVICE_COMPILE__) &&                                        \
    __has_builtin(__builtin_amdgcn_global_load_async_to_lds_b128) &&          \
    __has_builtin(__builtin_amdgcn_s_wait_asynccnt)
#define USE_ASYNC_COPY 1
#else
#define USE_ASYNC_COPY 0
#endif

union BFrag {
  v16bf v;
  v8bf  h[2];
  unsigned short u[16];
};

__device__ __forceinline__ unsigned short f2bf(float f) {
  union { float f; unsigned int u; } x;
  x.f = f;
  return (unsigned short)((x.u + 0x7fffu + ((x.u >> 16) & 1u)) >> 16);
}

__device__ __forceinline__ float geluf(float x) {
  return 0.5f * x * (1.0f + erff(x * 0.70710678118654752f));
}

__device__ __forceinline__ v8f wmma_bf16(v16bf a, v16bf b, v8f c) {
  // (neg_a, A, neg_b, B, c_mod, C, reuse_a, reuse_b)
  return __builtin_amdgcn_wmma_f32_16x16x32_bf16(false, a, false, b, (short)0, c,
                                                 false, false);
}

// A fragment: 16x32 bf16 from row-major [M x lda].
// lanes 0-15: row m0+lane, K={k0..k0+15} in e[0..15] (contiguous)
// lanes 16-31: row m0+lane-16, K={k0+16..k0+31}
__device__ __forceinline__ v16bf frag_a(const unsigned short* A, int lda, int m0,
                                        int k0, int lane, int maxRow) {
  int m = m0 + (lane & 15);
  if (m > maxRow) m = maxRow;
  int hi = (lane >> 4) & 1;
  const unsigned short* p = A + (size_t)m * lda + k0 + hi * 16;
  BFrag f;
  f.h[0] = *(const v8bf*)p;
  f.h[1] = *(const v8bf*)(p + 8);
  return f.v;
}

// B fragment for C = A * W^T, W row-major [N x ldw]; B[k][n] = W[n][k].
// lane: column n = n0 + (lane&15); K rows = k0 + (lane>>4)*16 .. +15 (contiguous)
__device__ __forceinline__ v16bf frag_bt(const unsigned short* W, int ldw, int n0,
                                         int k0, int lane, int maxRow) {
  int n = n0 + (lane & 15);
  if (n > maxRow) n = maxRow;
  int hi = (lane >> 4) & 1;
  const unsigned short* p = W + (size_t)n * ldw + k0 + hi * 16;
  BFrag f;
  f.h[0] = *(const v8bf*)p;
  f.h[1] = *(const v8bf*)(p + 8);
  return f.v;
}

// B fragment for C = A * V, V row-major [K x ldv] (per-lane gather; used on LDS).
__device__ __forceinline__ v16bf frag_b(const unsigned short* V, int ldv, int k0,
                                        int n0, int lane, int maxK) {
  int n = n0 + (lane & 15);
  int hi = (lane >> 4) & 1;
  BFrag f;
#pragma unroll
  for (int kk = 0; kk < 16; ++kk) {
    int k = k0 + hi * 16 + kk;
    if (k > maxK) k = maxK;
    f.u[kk] = V[(size_t)k * ldv + n];
  }
  return f.v;
}

// ---------------------------------------------------------------------------
// Model constants
// ---------------------------------------------------------------------------
#define DIMC   192
#define MLPC   768
#define NQ     4097
#define NKV    1025
#define SEQ    8
#define MTOK   (SEQ * NQ)    /* 32776 */
#define RTOK   (SEQ * NKV)   /* 8200  */
#define PTOK   (SEQ * 1024)  /* 8192  */

// ---------------------------------------------------------------------------
// Generic WMMA GEMM:  C[M,N] = A[M,K] * W[N,K]^T  (+bias, gelu, residual)
// mode bits: 1=write bf16 Cb, 2=accumulate into f32 Cf (residual),
//            4=gelu, 8=SR row remap for Cb (row -> row + row/1024 + 1)
// ---------------------------------------------------------------------------
__global__ __launch_bounds__(256) void gemm_kernel(
    const unsigned short* __restrict__ A, int lda,
    const unsigned short* __restrict__ W, int ldw,
    const float* __restrict__ bias,
    int M, int N, int K,
    unsigned short* __restrict__ Cb, int ldcb,
    float* __restrict__ Cf, int ldcf,
    int mode) {
  int lane = threadIdx.x & 31;
  int wave = threadIdx.x >> 5;
  int wm = wave & 3, wn = wave >> 2;                // 4x2 waves = 64x32 tile
  int m0 = blockIdx.x * 64 + wm * 16;
  int n0 = blockIdx.y * 32 + wn * 16;
  if (n0 >= N) return;

  v8f acc = {};
  for (int k0 = 0; k0 < K; k0 += 32) {
    if (k0 + 32 < K) {
      // next-tile prefetch (global_prefetch_b8)
      int mp = m0 + (lane & 15); if (mp > M - 1) mp = M - 1;
      __builtin_prefetch(A + (size_t)mp * lda + k0 + 32, 0, 3);
      __builtin_prefetch(W + (size_t)(n0 + (lane & 15)) * ldw + k0 + 32, 0, 3);
    }
    v16bf a = frag_a(A, lda, m0, k0, lane, M - 1);
    v16bf b = frag_bt(W, ldw, n0, k0, lane, N - 1);
    acc = wmma_bf16(a, b, acc);
  }

  int n = n0 + (lane & 15);
  int hi = (lane >> 4) & 1;
  float bv = bias ? bias[n] : 0.0f;
#pragma unroll
  for (int r = 0; r < 8; ++r) {
    int m = m0 + r + hi * 8;
    if (m >= M) continue;
    float v = acc[r] + bv;
    if (mode & 4) v = geluf(v);
    if (mode & 2) Cf[(size_t)m * ldcf + n] += v;
    if (mode & 1) {
      size_t row = (size_t)m;
      if (mode & 8) row = (size_t)m + (m >> 10) + 1;  // seq of 1024 -> slot 1..1024
      Cb[row * ldcb + n] = f2bf(v);
    }
  }
}

// ---------------------------------------------------------------------------
// LayerNorm over dim 192 (block = 192 threads, one token per block)
// ---------------------------------------------------------------------------
__global__ __launch_bounds__(192) void ln_kernel(
    const float* __restrict__ X, const float* __restrict__ g,
    const float* __restrict__ b, unsigned short* __restrict__ outB,
    float* __restrict__ outF) {
  __shared__ float part[6];
  __shared__ float smean, srstd;
  size_t tok = blockIdx.x;
  int d = threadIdx.x, wv = d >> 5;
  float v = X[tok * DIMC + d];
  float s = v;
#pragma unroll
  for (int o = 16; o; o >>= 1) s += __shfl_xor(s, o, 32);
  if ((d & 31) == 0) part[wv] = s;
  __syncthreads();
  if (d == 0) {
    float t = 0;
    for (int i = 0; i < 6; ++i) t += part[i];
    smean = t / 192.0f;
  }
  __syncthreads();
  float c = v - smean;
  s = c * c;
#pragma unroll
  for (int o = 16; o; o >>= 1) s += __shfl_xor(s, o, 32);
  if ((d & 31) == 0) part[wv] = s;
  __syncthreads();
  if (d == 0) {
    float t = 0;
    for (int i = 0; i < 6; ++i) t += part[i];
    srstd = rsqrtf(t / 192.0f + 1e-5f);
  }
  __syncthreads();
  float y = c * srstd * g[d] + b[d];
  if (outB) outB[tok * DIMC + d] = f2bf(y);
  if (outF) outF[tok * DIMC + d] = y;
}

// ---------------------------------------------------------------------------
// Patch embed (1x1x1 conv C=3 -> 192) + LN + cls token + pos emb -> X f32
// ---------------------------------------------------------------------------
__global__ __launch_bounds__(192) void patch_embed_kernel(
    const float* __restrict__ x, const float* __restrict__ pe_w,
    const float* __restrict__ pe_b, const float* __restrict__ lng,
    const float* __restrict__ lnb, const float* __restrict__ pos,
    const float* __restrict__ cls_tok, float* __restrict__ X) {
  __shared__ float part[6];
  __shared__ float smean, srstd;
  int sTok = blockIdx.x;
  int s = sTok / NQ, tok = sTok - s * NQ;
  int b = s >> 1, t = s & 1;
  int d = threadIdx.x;
  size_t orow = (size_t)sTok * DIMC;
  const float* pp = pos + ((size_t)t * NQ + tok) * DIMC;
  if (tok == 0) {
    X[orow + d] = cls_tok[d] + pp[d];
    return;
  }
  int p = tok - 1;
  float acc = pe_b[d];
#pragma unroll
  for (int c = 0; c < 3; ++c)
    acc += x[((size_t)(b * 3 + c) * 2 + t) * 4096 + p] * pe_w[d * 3 + c];
  int wv = d >> 5;
  float sum = acc;
#pragma unroll
  for (int o = 16; o; o >>= 1) sum += __shfl_xor(sum, o, 32);
  if ((d & 31) == 0) part[wv] = sum;
  __syncthreads();
  if (d == 0) {
    float tt = 0;
    for (int i = 0; i < 6; ++i) tt += part[i];
    smean = tt / 192.0f;
  }
  __syncthreads();
  float c0 = acc - smean;
  sum = c0 * c0;
#pragma unroll
  for (int o = 16; o; o >>= 1) sum += __shfl_xor(sum, o, 32);
  if ((d & 31) == 0) part[wv] = sum;
  __syncthreads();
  if (d == 0) {
    float tt = 0;
    for (int i = 0; i < 6; ++i) tt += part[i];
    srstd = rsqrtf(tt / 192.0f + 1e-5f);
  }
  __syncthreads();
  X[orow + d] = c0 * srstd * lng[d] + lnb[d] + pp[d];
}

// ---------------------------------------------------------------------------
// im2col for SR conv (2x2 stride 2): IM[(s*1024+p)*768 + c*4+i*2+j]
// ---------------------------------------------------------------------------
__global__ void im2col_kernel(const unsigned short* __restrict__ XN,
                              unsigned short* __restrict__ IM) {
  size_t idx = (size_t)blockIdx.x * 256 + threadIdx.x;
  const size_t total = (size_t)PTOK * MLPC;
  if (idx >= total) return;
  int k = (int)(idx % MLPC);
  size_t sp = idx / MLPC;
  int p = (int)(sp & 1023);
  int s = (int)(sp >> 10);
  int c = k >> 2, ij = k & 3, i = ij >> 1, j = ij & 1;
  int py = p >> 5, px = p & 31;
  int row = 1 + ((py * 2 + i) << 6) + px * 2 + j;
  IM[idx] = XN[((size_t)s * NQ + row) * DIMC + c];
}

// copy cls rows of XN into slot 0 of each reduced sequence
__global__ __launch_bounds__(192) void clscopy_kernel(
    const unsigned short* __restrict__ XN, unsigned short* __restrict__ XKB) {
  int s = blockIdx.x, d = threadIdx.x;
  XKB[(size_t)s * NKV * DIMC + d] = XN[(size_t)s * NQ * DIMC + d];
}

// ---------------------------------------------------------------------------
// Flash attention: per (q-tile of 128, head, seq). Q in QO (bf16), K/V packed
// in KV [RTOK x 384]. K/V tiles staged cooperatively into LDS (async copy on
// gfx1250 when available). O written in place over Q.
// ---------------------------------------------------------------------------
__global__ __launch_bounds__(256) void attn_kernel(
    unsigned short* __restrict__ QO, const unsigned short* __restrict__ KV) {
  __shared__ unsigned short kt[32 * 64];          // K tile [kv][dh]
  __shared__ unsigned short vt[32 * 64];          // V tile [kv][dh]
  __shared__ unsigned short pLds[8][16 * 32];     // per-wave P transpose
  int lane = threadIdx.x & 31, wave = threadIdx.x >> 5;
  int s = blockIdx.z, head = blockIdx.y;
  int q0 = blockIdx.x * 128 + wave * 16;

  unsigned short* Qp = QO + (size_t)s * NQ * DIMC + head * 64;
  const unsigned short* Kp = KV + (size_t)s * NKV * 384 + head * 64;
  const unsigned short* Vp = Kp + DIMC;

  v16bf aq0 = frag_a(Qp, DIMC, q0, 0, lane, NQ - 1);
  v16bf aq1 = frag_a(Qp, DIMC, q0, 32, lane, NQ - 1);

  v8f o0 = {}, o1 = {}, o2 = {}, o3 = {};
  float mx[8], li[8];
#pragma unroll
  for (int r = 0; r < 8; ++r) { mx[r] = -1e30f; li[r] = 0.0f; }
  int hi = (lane >> 4) & 1;
  int nl = lane & 15;
  const float scale = 0.125f;  // DH^-0.5

  int srow = threadIdx.x >> 3;         // 0..31 : kv row this thread stages
  int scol = (threadIdx.x & 7) << 3;   // 0..56 : dh chunk (8 bf16 = 16 B)

  for (int kv0 = 0; kv0 < NKV; kv0 += 32) {
    __syncthreads();  // previous tile fully consumed
    {
      int gr = kv0 + srow;
      if (gr >= NKV) gr = NKV - 1;
      const unsigned short* gk = Kp + (size_t)gr * 384 + scol;
      const unsigned short* gv = Vp + (size_t)gr * 384 + scol;
#if USE_ASYNC_COPY
      __builtin_amdgcn_global_load_async_to_lds_b128(
          (v4i_vs*)gk,
          (__attribute__((address_space(3))) v4i_vs*)&kt[srow * 64 + scol], 0, 0);
      __builtin_amdgcn_global_load_async_to_lds_b128(
          (v4i_vs*)gv,
          (__attribute__((address_space(3))) v4i_vs*)&vt[srow * 64 + scol], 0, 0);
      __builtin_amdgcn_s_wait_asynccnt(0);
#else
      *(v8bf*)&kt[srow * 64 + scol] = *(const v8bf*)gk;
      *(v8bf*)&vt[srow * 64 + scol] = *(const v8bf*)gv;
#endif
    }
    __syncthreads();  // tile visible to all waves

    v8f s0 = {}, s1 = {};
    s0 = wmma_bf16(aq0, frag_bt(kt, 64, 0, 0, lane, 31), s0);
    s0 = wmma_bf16(aq1, frag_bt(kt, 64, 0, 32, lane, 31), s0);
    s1 = wmma_bf16(aq0, frag_bt(kt, 64, 16, 0, lane, 31), s1);
    s1 = wmma_bf16(aq1, frag_bt(kt, 64, 16, 32, lane, 31), s1);

    int col0 = kv0 + nl, col1 = kv0 + 16 + nl;
    float alpha[8];
#pragma unroll
    for (int r = 0; r < 8; ++r) {
      float v0 = (col0 < NKV) ? s0[r] * scale : -1e30f;
      float v1 = (col1 < NKV) ? s1[r] * scale : -1e30f;
      float m = fmaxf(v0, v1);
#pragma unroll
      for (int off = 1; off < 16; off <<= 1) m = fmaxf(m, __shfl_xor(m, off, 32));
      float mn = fmaxf(mx[r], m);
      float a = __expf(mx[r] - mn);
      float p0 = __expf(v0 - mn);
      float p1 = __expf(v1 - mn);
      float rs = p0 + p1;
#pragma unroll
      for (int off = 1; off < 16; off <<= 1) rs += __shfl_xor(rs, off, 32);
      li[r] = li[r] * a + rs;
      mx[r] = mn;
      alpha[r] = a;
      s0[r] = p0;
      s1[r] = p1;
    }
#pragma unroll
    for (int r = 0; r < 8; ++r) {
      o0[r] *= alpha[r]; o1[r] *= alpha[r]; o2[r] *= alpha[r]; o3[r] *= alpha[r];
    }
    // C-layout -> LDS -> A-layout for P
    unsigned short* pl = pLds[wave];
#pragma unroll
    for (int r = 0; r < 8; ++r) {
      pl[(r + 8 * hi) * 32 + nl] = f2bf(s0[r]);
      pl[(r + 8 * hi) * 32 + 16 + nl] = f2bf(s1[r]);
    }
    v16bf ap = frag_a(pl, 32, 0, 0, lane, 15);
    o0 = wmma_bf16(ap, frag_b(vt, 64, 0, 0, lane, 31), o0);
    o1 = wmma_bf16(ap, frag_b(vt, 64, 0, 16, lane, 31), o1);
    o2 = wmma_bf16(ap, frag_b(vt, 64, 0, 32, lane, 31), o2);
    o3 = wmma_bf16(ap, frag_b(vt, 64, 0, 48, lane, 31), o3);
  }

#pragma unroll
  for (int r = 0; r < 8; ++r) {
    int m = q0 + r + 8 * hi;
    if (m >= NQ) continue;
    float inv = 1.0f / li[r];
    size_t base = ((size_t)s * NQ + m) * DIMC + head * 64 + nl;
    QO[base + 0]  = f2bf(o0[r] * inv);
    QO[base + 16] = f2bf(o1[r] * inv);
    QO[base + 32] = f2bf(o2[r] * inv);
    QO[base + 48] = f2bf(o3[r] * inv);
  }
}

// ---------------------------------------------------------------------------
// Temporal transformer: seq len 2, dim 192, 2 layers + final LN. One block
// per batch element (grid = 4 blocks, 192 threads).
// ---------------------------------------------------------------------------
__device__ __forceinline__ void ln2tok(float (*z)[DIMC], float (*zn)[DIMC],
                                       const float* g, const float* bb, int d) {
  __shared__ float part[2][6];
  __shared__ float mean[2], rstd[2];
  int wv = d >> 5, ln = d & 31;
  for (int t = 0; t < 2; ++t) {
    float s = z[t][d];
#pragma unroll
    for (int o = 16; o; o >>= 1) s += __shfl_xor(s, o, 32);
    if (ln == 0) part[t][wv] = s;
  }
  __syncthreads();
  if (d < 2) {
    float s = 0;
    for (int i = 0; i < 6; ++i) s += part[d][i];
    mean[d] = s / 192.0f;
  }
  __syncthreads();
  for (int t = 0; t < 2; ++t) {
    float c = z[t][d] - mean[t];
    float s = c * c;
#pragma unroll
    for (int o = 16; o; o >>= 1) s += __shfl_xor(s, o, 32);
    if (ln == 0) part[t][wv] = s;
  }
  __syncthreads();
  if (d < 2) {
    float s = 0;
    for (int i = 0; i < 6; ++i) s += part[d][i];
    rstd[d] = rsqrtf(s / 192.0f + 1e-5f);
  }
  __syncthreads();
  for (int t = 0; t < 2; ++t)
    zn[t][d] = (z[t][d] - mean[t]) * rstd[t] * g[d] + bb[d];
  __syncthreads();
}

__global__ __launch_bounds__(192) void temporal_kernel(
    const float* __restrict__ XF, const float* __restrict__ wqkv,
    const float* __restrict__ wo, const float* __restrict__ bo,
    const float* __restrict__ ln1g, const float* __restrict__ ln1b,
    const float* __restrict__ ln2g, const float* __restrict__ ln2b,
    const float* __restrict__ w1, const float* __restrict__ b1,
    const float* __restrict__ w2, const float* __restrict__ b2,
    const float* __restrict__ lnfg, const float* __restrict__ lnfb,
    float* __restrict__ Z) {
  __shared__ float z[2][DIMC], zn[2][DIMC], qkv[2][576], ao[2][DIMC], hh[2][MLPC];
  __shared__ float dots[3][2][2];
  int b = blockIdx.x, d = threadIdx.x;
  for (int t = 0; t < 2; ++t)
    z[t][d] = XF[((size_t)(b * 2 + t) * NQ) * DIMC + d];
  __syncthreads();

  for (int L = 0; L < 2; ++L) {
    ln2tok(z, zn, ln1g + L * DIMC, ln1b + L * DIMC, d);
    for (int t = 0; t < 2; ++t)
      for (int i = d; i < 576; i += 192) {
        const float* w = wqkv + ((size_t)L * 576 + i) * DIMC;
        float s = 0;
        for (int k = 0; k < DIMC; ++k) s += zn[t][k] * w[k];
        qkv[t][i] = s;
      }
    __syncthreads();
    if (d < 12) {
      int h = d >> 2, rem = d & 3, i = rem >> 1, j = rem & 1;
      float s = 0;
      for (int k = 0; k < 64; ++k)
        s += qkv[i][h * 64 + k] * qkv[j][DIMC + h * 64 + k];
      dots[h][i][j] = s * 0.125f;
    }
    __syncthreads();
    int h = d >> 6;
    for (int t = 0; t < 2; ++t) {
      float d0 = dots[h][t][0], d1 = dots[h][t][1];
      float m = fmaxf(d0, d1);
      float e0 = __expf(d0 - m), e1 = __expf(d1 - m);
      float inv = 1.0f / (e0 + e1);
      ao[t][d] = (e0 * qkv[0][384 + d] + e1 * qkv[1][384 + d]) * inv;
    }
    __syncthreads();
    for (int t = 0; t < 2; ++t) {
      const float* w = wo + ((size_t)L * DIMC + d) * DIMC;
      float s = 0;
      for (int k = 0; k < DIMC; ++k) s += ao[t][k] * w[k];
      z[t][d] += s + bo[L * DIMC + d];
    }
    __syncthreads();
    ln2tok(z, zn, ln2g + L * DIMC, ln2b + L * DIMC, d);
    for (int t = 0; t < 2; ++t)
      for (int i = d; i < MLPC; i += 192) {
        const float* w = w1 + ((size_t)L * MLPC + i) * DIMC;
        float s = 0;
        for (int k = 0; k < DIMC; ++k) s += zn[t][k] * w[k];
        hh[t][i] = geluf(s + b1[L * MLPC + i]);
      }
    __syncthreads();
    for (int t = 0; t < 2; ++t) {
      const float* w = w2 + ((size_t)L * DIMC + d) * MLPC;
      float s = 0;
      for (int k = 0; k < MLPC; ++k) s += hh[t][k] * w[k];
      z[t][d] += s + b2[L * DIMC + d];
    }
    __syncthreads();
  }
  ln2tok(z, zn, lnfg, lnfb, d);
  for (int t = 0; t < 2; ++t) Z[(size_t)(b * 2 + t) * DIMC + d] = zn[t][d];
}

// out = o * (1 + z)
__global__ void combine_kernel(const float* __restrict__ XF,
                               const float* __restrict__ Z,
                               float* __restrict__ out) {
  size_t idx = (size_t)blockIdx.x * 256 + threadIdx.x;
  const size_t total = (size_t)SEQ * 4096 * DIMC;
  if (idx >= total) return;
  int d = (int)(idx % DIMC);
  size_t pd = idx / DIMC;
  int p = (int)(pd & 4095);
  int s = (int)(pd >> 12);
  float o = XF[((size_t)s * NQ + 1 + p) * DIMC + d];
  out[idx] = o * (1.0f + Z[(size_t)s * DIMC + d]);
}

__global__ void f2bf_kernel(const float* __restrict__ in,
                            unsigned short* __restrict__ out, int n) {
  int i = blockIdx.x * 256 + threadIdx.x;
  if (i < n) out[i] = f2bf(in[i]);
}

// ---------------------------------------------------------------------------
// Host launcher
// ---------------------------------------------------------------------------
extern "C" void kernel_launch(void* const* d_in, const int* in_sizes, int n_in,
                              void* d_out, int out_size, void* d_ws,
                              size_t ws_size, hipStream_t stream) {
  const float* x        = (const float*)d_in[0];
  const float* pe_w     = (const float*)d_in[1];
  const float* pe_b     = (const float*)d_in[2];
  const float* pe_ln_g  = (const float*)d_in[3];
  const float* pe_ln_b  = (const float*)d_in[4];
  const float* pos_emb  = (const float*)d_in[5];
  const float* sp_tok   = (const float*)d_in[6];
  const float* s_ln1_g  = (const float*)d_in[7];
  const float* s_ln1_b  = (const float*)d_in[8];
  const float* s_wq     = (const float*)d_in[9];
  const float* s_wkv    = (const float*)d_in[10];
  const float* s_wo     = (const float*)d_in[11];
  const float* s_bo     = (const float*)d_in[12];
  const float* s_sr_w   = (const float*)d_in[13];
  const float* s_sr_b   = (const float*)d_in[14];
  const float* s_ln2_g  = (const float*)d_in[15];
  const float* s_ln2_b  = (const float*)d_in[16];
  const float* s_w1     = (const float*)d_in[17];
  const float* s_b1     = (const float*)d_in[18];
  const float* s_w2     = (const float*)d_in[19];
  const float* s_b2     = (const float*)d_in[20];
  const float* s_lnf_g  = (const float*)d_in[21];
  const float* s_lnf_b  = (const float*)d_in[22];
  const float* t_ln1_g  = (const float*)d_in[23];
  const float* t_ln1_b  = (const float*)d_in[24];
  const float* t_wqkv   = (const float*)d_in[25];
  const float* t_wo     = (const float*)d_in[26];
  const float* t_bo     = (const float*)d_in[27];
  const float* t_ln2_g  = (const float*)d_in[28];
  const float* t_ln2_b  = (const float*)d_in[29];
  const float* t_w1     = (const float*)d_in[30];
  const float* t_b1     = (const float*)d_in[31];
  const float* t_w2     = (const float*)d_in[32];
  const float* t_b2     = (const float*)d_in[33];
  const float* t_lnf_g  = (const float*)d_in[34];
  const float* t_lnf_b  = (const float*)d_in[35];
  float* out = (float*)d_out;

  char* ws = (char*)d_ws;
  size_t off = 0;
  auto alloc = [&](size_t bytes) {
    size_t o = off;
    off += (bytes + 255) & ~(size_t)255;
    return o;
  };
  float* X            = (float*)(ws + alloc((size_t)MTOK * DIMC * 4));
  unsigned short* XN  = (unsigned short*)(ws + alloc((size_t)MTOK * DIMC * 2));
  unsigned short* QB  = (unsigned short*)(ws + alloc((size_t)MTOK * DIMC * 2));
  unsigned short* KVB = (unsigned short*)(ws + alloc((size_t)RTOK * 384 * 2));
  unsigned short* XKB = (unsigned short*)(ws + alloc((size_t)RTOK * DIMC * 2));
  unsigned short* WB  = (unsigned short*)(ws + alloc((size_t)1179648 * 2));
  char* BIG           = ws + alloc((size_t)MTOK * MLPC * 2);
  float* ZB           = (float*)(ws + alloc(8 * DIMC * 4));
  // Time-multiplexed aliases of BIG (never live simultaneously):
  unsigned short* IM = (unsigned short*)BIG;  // im2col [PTOK x 768]
  unsigned short* HB = (unsigned short*)BIG;  // FF hidden [MTOK x 768]
  float* XF          = (float*)BIG;           // final LN f32 [MTOK x 192]

  unsigned short* WQb  = WB + 0;
  unsigned short* WKVb = WB + 73728;
  unsigned short* WOb  = WB + 221184;
  unsigned short* SRWb = WB + 294912;
  unsigned short* W1b  = WB + 589824;
  unsigned short* W2b  = WB + 884736;

  auto cvt = [&](const float* src, unsigned short* dst, int n) {
    f2bf_kernel<<<(n + 255) / 256, 256, 0, stream>>>(src, dst, n);
  };
  cvt(s_wq, WQb, 73728);
  cvt(s_wkv, WKVb, 147456);
  cvt(s_wo, WOb, 73728);
  cvt(s_sr_w, SRWb, 294912);
  cvt(s_w1, W1b, 294912);
  cvt(s_w2, W2b, 294912);

  patch_embed_kernel<<<SEQ * NQ, 192, 0, stream>>>(x, pe_w, pe_b, pe_ln_g,
                                                   pe_ln_b, pos_emb, sp_tok, X);

  const int MT = (MTOK + 63) / 64;   // 513
  const int RT = (RTOK + 63) / 64;   // 129

  for (int L = 0; L < 2; ++L) {
    // LN1 -> XN (bf16)
    ln_kernel<<<MTOK, 192, 0, stream>>>(X, s_ln1_g + L * DIMC, s_ln1_b + L * DIMC,
                                        XN, nullptr);
    // Q = XN * Wq^T
    gemm_kernel<<<dim3(MT, 6), 256, 0, stream>>>(
        XN, DIMC, WQb + L * 36864, DIMC, nullptr, MTOK, DIMC, DIMC, QB, DIMC,
        nullptr, 0, 1);
    // reduced token set: cls + SR conv (as im2col GEMM)
    clscopy_kernel<<<SEQ, 192, 0, stream>>>(XN, XKB);
    im2col_kernel<<<24576, 256, 0, stream>>>(XN, IM);
    gemm_kernel<<<dim3(PTOK / 64, 6), 256, 0, stream>>>(
        IM, MLPC, SRWb + L * 147456, MLPC, s_sr_b + L * DIMC, PTOK, DIMC, MLPC,
        XKB, DIMC, nullptr, 0, 1 | 8);
    // KV = XKB * Wkv^T
    gemm_kernel<<<dim3(RT, 12), 256, 0, stream>>>(
        XKB, DIMC, WKVb + L * 73728, DIMC, nullptr, RTOK, 384, DIMC, KVB, 384,
        nullptr, 0, 1);
    // attention (O overwrites Q)
    attn_kernel<<<dim3(33, 3, SEQ), 256, 0, stream>>>(QB, KVB);
    // X += O * Wo^T + bo
    gemm_kernel<<<dim3(MT, 6), 256, 0, stream>>>(
        QB, DIMC, WOb + L * 36864, DIMC, s_bo + L * DIMC, MTOK, DIMC, DIMC,
        nullptr, 0, X, DIMC, 2);
    // LN2 -> XN
    ln_kernel<<<MTOK, 192, 0, stream>>>(X, s_ln2_g + L * DIMC, s_ln2_b + L * DIMC,
                                        XN, nullptr);
    // H = gelu(XN * W1^T + b1)
    gemm_kernel<<<dim3(MT, 24), 256, 0, stream>>>(
        XN, DIMC, W1b + L * 147456, DIMC, s_b1 + L * MLPC, MTOK, MLPC, DIMC, HB,
        MLPC, nullptr, 0, 1 | 4);
    // X += H * W2^T + b2
    gemm_kernel<<<dim3(MT, 6), 256, 0, stream>>>(
        HB, MLPC, W2b + L * 147456, MLPC, s_b2 + L * DIMC, MTOK, DIMC, MLPC,
        nullptr, 0, X, DIMC, 2);
  }

  // final LN -> XF (f32)
  ln_kernel<<<MTOK, 192, 0, stream>>>(X, s_lnf_g, s_lnf_b, nullptr, XF);
  // temporal transformer on cls tokens -> ZB
  temporal_kernel<<<4, 192, 0, stream>>>(XF, t_wqkv, t_wo, t_bo, t_ln1_g, t_ln1_b,
                                         t_ln2_g, t_ln2_b, t_w1, t_b1, t_w2, t_b2,
                                         t_lnf_g, t_lnf_b, ZB);
  // out = patches * (1 + z)
  combine_kernel<<<24576, 256, 0, stream>>>(XF, ZB, out);
}
